// ST_BLOCK_7_82884278878474
// MI455X (gfx1250) — matrix operand order repaired and checked by reading, not compiled, not run on hardware
//
#include <hip/hip_runtime.h>
#include <hip/hip_bf16.h>
#include <stdint.h>

// ---------------------------------------------------------------------------
// Shapes
// ---------------------------------------------------------------------------
constexpr int BB   = 16;      // batch
constexpr int CIN  = 32;
constexpr int CO   = 64;
constexpr int NN   = 1024;    // nodes
constexpr int TT   = 24;      // time
constexpr int TP   = TT + 2;  // padded time for conv_t
constexpr long NBT = (long)NN * BB * TT;  // 393216 rows, layout row=(n,b,t)

typedef __bf16 bf16;
typedef __attribute__((ext_vector_type(16))) __bf16 v16bf;
typedef __attribute__((ext_vector_type(8)))  float  v8f;

union Frag { v16bf v; uint32_t u[8]; __bf16 h[16]; };

static __device__ __forceinline__ v8f wmma_bf16(v16bf a, v16bf b, v8f c) {
  // v_wmma_f32_16x16x32_bf16: (neg_a, A, neg_b, B, c_mod, C, reuse_a, reuse_b)
  return __builtin_amdgcn_wmma_f32_16x16x32_bf16(false, a, false, b, (short)0, c, false, false);
}

// A-fragment (16x32 bf16, M x K): lane ln holds row m; element e -> k = e + (hi?16:0)
static __device__ __forceinline__ void ld_afrag(Frag& f, const bf16* rowbase, int hi) {
#pragma unroll
  for (int j = 0; j < 8; ++j) {
    int k = 2 * j + (hi ? 16 : 0);
    f.u[j] = *(const uint32_t*)(rowbase + k);
  }
}
// B-fragment (32x16 bf16, K x N): lane ln holds col n; element e -> k = e + (hi?16:0)
static __device__ __forceinline__ void ld_bfrag(Frag& f, const bf16* colbase, int hi) {
#pragma unroll
  for (int j = 0; j < 8; ++j) {
    int k = 2 * j + (hi ? 16 : 0);
    f.u[j] = *(const uint32_t*)(colbase + k);
  }
}

// ---- gfx1250 async global->LDS staging (ASYNCcnt path), with safe fallback --
#if defined(__has_builtin)
#if __has_builtin(__builtin_amdgcn_global_load_async_to_lds_b128)
#define USE_ASYNC_LDS 1
#endif
#endif

#ifdef USE_ASYNC_LDS
typedef int v4i_vs __attribute__((vector_size(16)));                 // gcc-style int4
typedef __attribute__((address_space(1))) v4i_vs* gp_b128;           // global AS1
typedef __attribute__((address_space(3))) v4i_vs* lp_b128;           // LDS AS3

static __device__ __forceinline__ void async_ld16(const bf16* gsrc, bf16* ldst) {
  __builtin_amdgcn_global_load_async_to_lds_b128((gp_b128)(gsrc), (lp_b128)(ldst), 0, 0);
}
static __device__ __forceinline__ void async_wait0() {
#if __has_builtin(__builtin_amdgcn_s_wait_asynccnt)
  __builtin_amdgcn_s_wait_asynccnt(0);
#else
  asm volatile("s_wait_asynccnt 0x0" ::: "memory");
#endif
}
#endif

// ---------------------------------------------------------------------------
// Generic bf16 WMMA GEMM:  C[M,Nc] = A[M,K] * B[K,Nc]  (+ epilogue)
// Block = 256 thr = 8 waves (4x2). Tile 128(M) x BN(N), K step 32.
// BN=64:  wave computes 32x32 (4 wmma / K-step)
// BN=128: wave computes 32x64 (8 wmma / K-step)  -> used for the heavy GEMMs
// AMODE 0: A row r at A + r*lda.  AMODE 1 (conv_t): row r=(nb,t) window into
//          padded XD, offset (nb*TP + t)*64, K=192 contiguous.
// ---------------------------------------------------------------------------
constexpr int EPI_BIAS_BF16       = 0;  // Cb = bf16(acc + bias[col])
constexpr int EPI_BIAS_DELAY_BF16 = 1;  // Cb = bf16((acc+bias[col]) * extra[node*T+t])
constexpr int EPI_BIAS_F32        = 2;  // Cf = acc + bias[col]
constexpr int EPI_NONE_BF16       = 3;  // Cb = bf16(acc)
constexpr int EPI_L2              = 4;  // Cb = bf16(2*acc - (row==col))
constexpr int EPI_ADD_F32         = 5;  // Cf += acc

template <int AMODE, int EPI, int BN>
__global__ __launch_bounds__(256) void gemm_bf16_wmma(
    const bf16* __restrict__ A, const bf16* __restrict__ Bm,
    const float* __restrict__ bias, const float* __restrict__ extra,
    float* __restrict__ Cf, bf16* __restrict__ Cb,
    int M, int Kdim, int Nc, int lda, int ldb, int ldc)
{
  constexpr int SN = BN / 32;          // per-wave 16x16 col subtiles
  __shared__ bf16 lsA[128 * 40];       // stride 40 -> conflict-free frag reads
  __shared__ bf16 lsB[BN * 40];        // stored transposed [n][k]

  const int tid = threadIdx.x;
  const int wid = tid >> 5, lane = tid & 31;
  const int hi = lane >> 4, ln = lane & 15;
  const long m0 = (long)blockIdx.y * 128;
  const long n0 = (long)blockIdx.x * BN;
  const int wm = (wid >> 1) * 32, wn = (wid & 1) * (BN / 2);

  v8f acc[2][SN];
#pragma unroll
  for (int i = 0; i < 2; ++i)
#pragma unroll
    for (int j = 0; j < SN; ++j) acc[i][j] = (v8f){0.f,0.f,0.f,0.f,0.f,0.f,0.f,0.f};

  // Per-thread tile-load coordinates
  const int ar   = tid >> 1;           // A: row 0..127
  const int acol = (tid & 1) * 16;     // A: col 0 or 16 (x16 bf16)
  long aoff;
  {
    long gr = m0 + ar;
    if (AMODE == 0) aoff = gr * (long)lda;
    else { long nb = gr / TT; long t = gr % TT; aoff = (nb * TP + t) * 64; }
  }
  const int bkr  = tid >> 3;           // B: k-row 0..31 (8 threads per row)
  const int bcol = (tid & 7) * (BN / 8);

  for (int k0 = 0; k0 < Kdim; k0 += 32) {
    // ---- stage A tile (128x32) ----
#ifdef USE_ASYNC_LDS
    async_ld16(A + aoff + k0 + acol,     &lsA[ar * 40 + acol]);
    async_ld16(A + aoff + k0 + acol + 8, &lsA[ar * 40 + acol + 8]);
#else
    uint4 va0 = *(const uint4*)(A + aoff + k0 + acol);
    uint4 va1 = *(const uint4*)(A + aoff + k0 + acol + 8);
    *(uint4*)(&lsA[ar * 40 + acol])     = va0;
    *(uint4*)(&lsA[ar * 40 + acol + 8]) = va1;
#endif
    // ---- stage B tile (32xBN), transposed into lsB[n][k] ----
#pragma unroll
    for (int i = 0; i < BN / 64; ++i) {
      union { uint4 u; bf16 h[8]; } vb;
      vb.u = *(const uint4*)(Bm + (long)(k0 + bkr) * ldb + n0 + bcol + i * 8);
#pragma unroll
      for (int j = 0; j < 8; ++j) lsB[(bcol + i * 8 + j) * 40 + bkr] = vb.h[j];
    }
#ifdef USE_ASYNC_LDS
    async_wait0();
#endif
    __syncthreads();

    Frag a0, a1, bfr[SN];
    ld_afrag(a0, &lsA[(wm + ln) * 40], hi);
    ld_afrag(a1, &lsA[(wm + 16 + ln) * 40], hi);
#pragma unroll
    for (int sn = 0; sn < SN; ++sn) ld_bfrag(bfr[sn], &lsB[(wn + sn * 16 + ln) * 40], hi);

#pragma unroll
    for (int sn = 0; sn < SN; ++sn) {
      acc[0][sn] = wmma_bf16(a0.v, bfr[sn].v, acc[0][sn]);
      acc[1][sn] = wmma_bf16(a1.v, bfr[sn].v, acc[1][sn]);
    }
    __syncthreads();
  }

  // ---- epilogue: C tile layout: lane ln = col, VGPR v -> row v + (hi?8:0) ----
#pragma unroll
  for (int sm = 0; sm < 2; ++sm)
#pragma unroll
    for (int sn = 0; sn < SN; ++sn)
#pragma unroll
      for (int v = 0; v < 8; ++v) {
        long row = m0 + wm + sm * 16 + (hi ? 8 : 0) + v;
        long col = n0 + wn + sn * 16 + ln;
        float val = acc[sm][sn][v];
        long idx = row * (long)ldc + col;
        if (EPI == EPI_BIAS_BF16) {
          Cb[idx] = (bf16)(val + bias[col]);
        } else if (EPI == EPI_BIAS_DELAY_BF16) {
          long node = row / (BB * TT); long t = row % TT;
          Cb[idx] = (bf16)((val + bias[col]) * extra[node * TT + t]);
        } else if (EPI == EPI_BIAS_F32) {
          Cf[idx] = val + bias[col];
        } else if (EPI == EPI_NONE_BF16) {
          Cb[idx] = (bf16)val;
        } else if (EPI == EPI_L2) {
          Cb[idx] = (bf16)(2.f * val - (row == col ? 1.f : 0.f));
        } else if (EPI == EPI_ADD_F32) {
          Cf[idx] += val;
        }
      }
}

// ---------------------------------------------------------------------------
// Cross-attention over s,t (24x24, C=64) per (n,b); one wave per (n,b).
// A = softmax(Q Kx^T / 8); out = A V  [+ x_input1].  T padded 24->32 tiles.
// ---------------------------------------------------------------------------
template <int ADD_X1, int PAD>
__global__ __launch_bounds__(256) void attn_kernel(
    const bf16* __restrict__ Q, const bf16* __restrict__ Kx,
    const bf16* __restrict__ V, const float* __restrict__ x1f,
    bf16* __restrict__ Out)
{
  __shared__ bf16 lsP[8][32 * 34];  // softmax probs per wave (stride 34)
  __shared__ bf16 lsV[8][24 * 64];  // staged V rows per wave

  const int tid = threadIdx.x;
  const int w = tid >> 5, lane = tid & 31;
  const int hi = lane >> 4, ln = lane & 15;
  const long nb = (long)blockIdx.x * 8 + w;   // (n,b) pair
  const long r0 = nb * TT;

  // stage V (24x64)
#pragma unroll
  for (int i = 0; i < 6; ++i) {
    int linear = i * 256 + lane * 8;
    int s = linear >> 6, c = linear & 63;
    *(uint4*)&lsV[w][s * 64 + c] = *(const uint4*)(V + (r0 + s) * 64 + c);
  }

  // Q a-fragments: [t-tile][k-chunk]
  Frag aq[2][2];
#pragma unroll
  for (int sm = 0; sm < 2; ++sm) {
    int m = sm * 16 + ln;
    bool valid = m < TT;
#pragma unroll
    for (int kc = 0; kc < 2; ++kc) {
      const bf16* base = Q + (r0 + m) * 64 + kc * 32;
#pragma unroll
      for (int j = 0; j < 8; ++j) {
        int k = 2 * j + (hi ? 16 : 0);
        aq[sm][kc].u[j] = valid ? *(const uint32_t*)(base + k) : 0u;
      }
    }
  }
  // Kx^T b-fragments: B[k=c][n=s] -> contiguous 16 bf16 per lane
  Frag bk[2][2];
#pragma unroll
  for (int sn = 0; sn < 2; ++sn) {
    int s = sn * 16 + ln;
    bool valid = s < TT;
#pragma unroll
    for (int kc = 0; kc < 2; ++kc) {
      if (valid) {
        const uint4* p = (const uint4*)(Kx + (r0 + s) * 64 + kc * 32 + (hi ? 16 : 0));
        ((uint4*)bk[sn][kc].u)[0] = p[0];
        ((uint4*)bk[sn][kc].u)[1] = p[1];
      } else {
#pragma unroll
        for (int j = 0; j < 8; ++j) bk[sn][kc].u[j] = 0u;
      }
    }
  }

  // scores
  v8f sc[2][2];
#pragma unroll
  for (int sm = 0; sm < 2; ++sm)
#pragma unroll
    for (int sn = 0; sn < 2; ++sn) {
      sc[sm][sn] = (v8f){0.f,0.f,0.f,0.f,0.f,0.f,0.f,0.f};
      sc[sm][sn] = wmma_bf16(aq[sm][0].v, bk[sn][0].v, sc[sm][sn]);
      sc[sm][sn] = wmma_bf16(aq[sm][1].v, bk[sn][1].v, sc[sm][sn]);
    }

  // softmax over s per row t (row lives across the 16 lanes of a half)
#pragma unroll
  for (int sm = 0; sm < 2; ++sm)
#pragma unroll
    for (int v = 0; v < 8; ++v) {
      int m = sm * 16 + (hi ? 8 : 0) + v;
      float v0 = sc[sm][0][v] * 0.125f;         // s = ln
      float v1 = sc[sm][1][v] * 0.125f;         // s = 16+ln
      int s1 = 16 + ln;
      bool ok1 = s1 < TT;
      float mx = fmaxf(v0, ok1 ? v1 : -3.0e38f);
#pragma unroll
      for (int off = 1; off < 16; off <<= 1) mx = fmaxf(mx, __shfl_xor(mx, off, 32));
      float e0 = __expf(v0 - mx);
      float e1 = ok1 ? __expf(v1 - mx) : 0.f;
      float sum = e0 + e1;
#pragma unroll
      for (int off = 1; off < 16; off <<= 1) sum += __shfl_xor(sum, off, 32);
      float inv = 1.f / sum;
      float p0 = e0 * inv, p1 = e1 * inv;
      if (m >= TT) { p0 = 0.f; p1 = 0.f; }
      lsP[w][m * 34 + ln] = (bf16)p0;
      lsP[w][m * 34 + s1] = (bf16)p1;
    }
  __syncthreads();

  // A-fragments of probs, B-fragments of V; out = P @ V
  Frag ap[2];
#pragma unroll
  for (int sm = 0; sm < 2; ++sm) ld_afrag(ap[sm], &lsP[w][(sm * 16 + ln) * 34], hi);

  Frag bv[4];
#pragma unroll
  for (int ni = 0; ni < 4; ++ni) {
    int c = ni * 16 + ln;
#pragma unroll
    for (int e = 0; e < 16; ++e) {
      int s = e + (hi ? 16 : 0);
      bv[ni].h[e] = (s < TT) ? lsV[w][s * 64 + c] : (bf16)0.f;
    }
  }

  v8f ov[2][4];
#pragma unroll
  for (int sm = 0; sm < 2; ++sm)
#pragma unroll
    for (int ni = 0; ni < 4; ++ni) {
      ov[sm][ni] = (v8f){0.f,0.f,0.f,0.f,0.f,0.f,0.f,0.f};
      ov[sm][ni] = wmma_bf16(ap[sm].v, bv[ni].v, ov[sm][ni]);
    }

  // store rows t < 24
#pragma unroll
  for (int sm = 0; sm < 2; ++sm)
#pragma unroll
    for (int ni = 0; ni < 4; ++ni)
#pragma unroll
      for (int v = 0; v < 8; ++v) {
        int t = sm * 16 + (hi ? 8 : 0) + v;
        if (t < TT) {
          int c = ni * 16 + ln;
          long row = r0 + t;
          float val = ov[sm][ni][v];
          if (ADD_X1) val += x1f[row * 64 + c];
          long didx = PAD ? ((nb * TP + t + 1) * 64 + c) : (row * 64 + c);
          Out[didx] = (bf16)val;
        }
      }
}

// ---------------------------------------------------------------------------
// Small prep / elementwise kernels
// ---------------------------------------------------------------------------
__global__ __launch_bounds__(256) void k_delay(const float* CD, const float* Wd, float* delay) {
  int idx = blockIdx.x * 256 + threadIdx.x;
  if (idx >= NN * TT) return;
  int n = idx / TT, t = idx % TT;
  float a = 0.f;
#pragma unroll
  for (int d = 0; d < 16; ++d) a += CD[n * 16 + d] * Wd[d * TT + t];
  float th = tanhf(a);
  delay[idx] = 1.f / (1.f + __expf(-th));
}

__global__ __launch_bounds__(256) void k_G(const float* CS, const float* Wmii, float* G) {
  int idx = blockIdx.x * 256 + threadIdx.x;
  if (idx >= NN * CO) return;
  int n = idx / CO, o = idx % CO;
  float a = 0.f;
#pragma unroll
  for (int e = 0; e < 16; ++e) a += CS[n * 16 + e] * Wmii[o * 16 + e];
  G[idx] = a;
}

__global__ __launch_bounds__(256) void k_adjsup(const float* adj, const float* sup,
                                               bf16* adjI, bf16* supb) {
  int idx = blockIdx.x * 256 + threadIdx.x;
  if (idx >= NN * NN) return;
  int r = idx / NN, c = idx % NN;
  adjI[idx] = (bf16)(adj[idx] + (r == c ? 1.f : 0.f));
  supb[idx] = (bf16)sup[idx];
}

__global__ __launch_bounds__(256) void k_packWT(const float* src, bf16* dst, int K, int Co) {
  int idx = blockIdx.x * 256 + threadIdx.x;
  if (idx >= K * Co) return;
  int k = idx / Co, o = idx % Co;
  dst[idx] = (bf16)src[o * K + k];
}

__global__ __launch_bounds__(256) void k_packWct(const float* Wc1, bf16* dst) {
  int idx = blockIdx.x * 256 + threadIdx.x;   // 192*64
  if (idx >= 192 * 64) return;
  int o = idx % 64, c = (idx / 64) % 64, kt = idx / 4096;
  dst[(kt * 64 + c) * 64 + o] = (bf16)Wc1[(o * 64 + c) * 3 + kt];
}

__global__ __launch_bounds__(256) void k_packWg(const float* Wg, bf16* dst) {
  int idx = blockIdx.x * 256 + threadIdx.x;   // 3*64*128
  if (idx >= 3 * 64 * 128) return;
  int k = idx / 8192, r = idx % 8192, c = r / 128, o = r % 128;
  dst[k * 8192 + c * 128 + o] = (bf16)Wg[o * 192 + c * 3 + k];
}

__global__ __launch_bounds__(256) void k_transpose_x(const float* x, bf16* xT) {
  long idx = (long)blockIdx.x * 256 + threadIdx.x;
  if (idx >= NBT * CIN) return;
  int c = idx & 31; long r = idx >> 5;
  int t = r % TT; long nb = r / TT; int b = nb & 15; int n = (int)(nb >> 4);
  xT[idx] = (bf16)x[(((long)b * CIN + c) * NN + n) * TT + t];
}

__global__ __launch_bounds__(256) void k_u(const bf16* xT, const float* Wpro,
                                           const float* bpro, float* u) {
  long idx = (long)blockIdx.x * 256 + threadIdx.x;
  if (idx >= NBT) return;
  float a = bpro[0];
#pragma unroll
  for (int c = 0; c < CIN; ++c) a += (float)xT[idx * CIN + c] * Wpro[c];
  u[idx] = a;
}

__global__ __launch_bounds__(256) void k_h(const float* u, const float* G,
                                           const float* bmii, bf16* h) {
  long idx = (long)blockIdx.x * 256 + threadIdx.x;
  if (idx >= NBT * CO) return;
  int o = idx & 63; long r = idx >> 6; long node = r / (BB * TT);
  h[idx] = (bf16)(u[r] * G[node * CO + o] + bmii[o]);
}

__global__ __launch_bounds__(256) void k_final(const float* x2, const float* x1f, float* out) {
  long idx = (long)blockIdx.x * 256 + threadIdx.x;
  if (idx >= (long)BB * CO * NN * TT) return;
  int t = idx % TT;
  int n = (idx / TT) % NN;
  int o = (idx / ((long)TT * NN)) % CO;
  int b = (int)(idx / ((long)TT * NN * CO));
  long r = ((long)n * BB + b) * TT + t;
  float f = x2[r * 128 + o] + x1f[r * 64 + o];
  float g = x2[r * 128 + 64 + o];
  out[idx] = f * (1.f / (1.f + __expf(-g)));
}

// ---------------------------------------------------------------------------
// Workspace layout (bytes). Liveness-based reuse: Z1<-D, Z2<-h, x2<-Q/K/V/XDpad.
// ---------------------------------------------------------------------------
constexpr size_t OFF_DELAY = 0;
constexpr size_t OFF_G     = OFF_DELAY + 98304;
constexpr size_t OFF_U     = OFF_G + 262144;
constexpr size_t OFF_SUPB  = OFF_U + 1572864;
constexpr size_t OFF_ADJI  = OFF_SUPB + 2097152;
constexpr size_t OFF_L2    = OFF_ADJI + 2097152;
constexpr size_t OFF_WDET  = OFF_L2 + 2097152;
constexpr size_t OFF_W1T   = OFF_WDET + 4096;
constexpr size_t OFF_WQTT  = OFF_W1T + 4096;
constexpr size_t OFF_WKTT  = OFF_WQTT + 4096;
constexpr size_t OFF_WVTT  = OFF_WKTT + 8192;
constexpr size_t OFF_WQST  = OFF_WVTT + 8192;
constexpr size_t OFF_WKST  = OFF_WQST + 8192;
constexpr size_t OFF_WVST  = OFF_WKST + 8192;
constexpr size_t OFF_WCT   = OFF_WVST + 8192;
constexpr size_t OFF_WG    = OFF_WCT + 24576;
constexpr size_t OFF_XT    = OFF_WG + 49152;
constexpr size_t OFF_X1F   = OFF_XT + 25165824;    // x_input1 f32 (live to end)
constexpr size_t OFF_D     = OFF_X1F + 100663296;  // later reused as Z1
constexpr size_t OFF_H     = OFF_D + 50331648;     // later reused as Z2
constexpr size_t OFF_S     = OFF_H + 50331648;
constexpr size_t OFF_Q     = OFF_S + 50331648;     // x2 (f32, 201MB) reuses Q..XDpad
constexpr size_t OFF_K     = OFF_Q + 50331648;
constexpr size_t OFF_V     = OFF_K + 50331648;
constexpr size_t OFF_XDP   = OFF_V + 50331648;     // padded XD, 54525952 B
constexpr size_t OFF_X1B   = OFF_XDP + 54525952;
constexpr size_t OFF_XS    = OFF_X1B + 50331648;
constexpr size_t WS_NEED   = OFF_XS + 50331648;    // ~564 MB
constexpr size_t OFF_X2    = OFF_Q;                // 201326592 B fits in Q..XDpad span

extern "C" void kernel_launch(void* const* d_in, const int* in_sizes, int n_in,
                              void* d_out, int out_size, void* d_ws, size_t ws_size,
                              hipStream_t stream) {
  if (ws_size < WS_NEED) return;
  char* W = (char*)d_ws;

  const float* x    = (const float*)d_in[0];
  const float* sup  = (const float*)d_in[1];
  const float* CD   = (const float*)d_in[2];
  const float* CS   = (const float*)d_in[3];
  const float* adj  = (const float*)d_in[4];
  const float* W_de = (const float*)d_in[5];  const float* b_de = (const float*)d_in[6];
  const float* Wd   = (const float*)d_in[7];
  const float* Wpro = (const float*)d_in[8];  const float* bpro = (const float*)d_in[9];
  const float* Wmii = (const float*)d_in[10]; const float* bmii = (const float*)d_in[11];
  const float* W_1  = (const float*)d_in[12]; const float* b_1  = (const float*)d_in[13];
  const float* Wq_t = (const float*)d_in[14]; const float* bq_t = (const float*)d_in[15];
  const float* Wk_t = (const float*)d_in[16]; const float* bk_t = (const float*)d_in[17];
  const float* Wv_t = (const float*)d_in[18]; const float* bv_t = (const float*)d_in[19];
  const float* W_c1 = (const float*)d_in[20]; const float* b_c1 = (const float*)d_in[21];
  const float* Wq_s = (const float*)d_in[22]; const float* bq_s = (const float*)d_in[23];
  const float* Wk_s = (const float*)d_in[24]; const float* bk_s = (const float*)d_in[25];
  const float* Wv_s = (const float*)d_in[26]; const float* bv_s = (const float*)d_in[27];
  const float* W_g  = (const float*)d_in[28]; const float* b_g  = (const float*)d_in[29];

  float* delay = (float*)(W + OFF_DELAY);
  float* G     = (float*)(W + OFF_G);
  float* u     = (float*)(W + OFF_U);
  bf16* supb   = (bf16*)(W + OFF_SUPB);
  bf16* adjI   = (bf16*)(W + OFF_ADJI);
  bf16* L2b    = (bf16*)(W + OFF_L2);
  bf16* wdeT = (bf16*)(W + OFF_WDET), *w1T = (bf16*)(W + OFF_W1T), *wqtT = (bf16*)(W + OFF_WQTT);
  bf16* wktT = (bf16*)(W + OFF_WKTT), *wvtT = (bf16*)(W + OFF_WVTT);
  bf16* wqsT = (bf16*)(W + OFF_WQST), *wksT = (bf16*)(W + OFF_WKST), *wvsT = (bf16*)(W + OFF_WVST);
  bf16* wct  = (bf16*)(W + OFF_WCT);
  bf16* wg   = (bf16*)(W + OFF_WG);
  bf16* xT   = (bf16*)(W + OFF_XT);
  float* x1f = (float*)(W + OFF_X1F);
  bf16* Db   = (bf16*)(W + OFF_D);
  bf16* hb   = (bf16*)(W + OFF_H);
  bf16* Sb   = (bf16*)(W + OFF_S);
  bf16* Qb   = (bf16*)(W + OFF_Q);
  bf16* Kb   = (bf16*)(W + OFF_K);
  bf16* Vb   = (bf16*)(W + OFF_V);
  bf16* XDp  = (bf16*)(W + OFF_XDP);
  bf16* x1b  = (bf16*)(W + OFF_X1B);
  bf16* XSb  = (bf16*)(W + OFF_XS);
  bf16* Z1b  = Db;   // reuse
  bf16* Z2b  = hb;   // reuse
  float* x2f = (float*)(W + OFF_X2);  // reuses Q/K/V/XDpad span
  float* out = (float*)d_out;

  // ---- prep ----
  k_delay<<<96, 256, 0, stream>>>(CD, Wd, delay);
  k_G<<<256, 256, 0, stream>>>(CS, Wmii, G);
  k_adjsup<<<4096, 256, 0, stream>>>(adj, sup, adjI, supb);
  k_packWT<<<8, 256, 0, stream>>>(W_de, wdeT, 32, 64);
  k_packWT<<<8, 256, 0, stream>>>(W_1,  w1T,  32, 64);
  k_packWT<<<8, 256, 0, stream>>>(Wq_t, wqtT, 32, 64);
  k_packWT<<<16, 256, 0, stream>>>(Wk_t, wktT, 64, 64);
  k_packWT<<<16, 256, 0, stream>>>(Wv_t, wvtT, 64, 64);
  k_packWT<<<16, 256, 0, stream>>>(Wq_s, wqsT, 64, 64);
  k_packWT<<<16, 256, 0, stream>>>(Wk_s, wksT, 64, 64);
  k_packWT<<<16, 256, 0, stream>>>(Wv_s, wvsT, 64, 64);
  k_packWct<<<48, 256, 0, stream>>>(W_c1, wct);
  k_packWg<<<96, 256, 0, stream>>>(W_g, wg);
  k_transpose_x<<<49152, 256, 0, stream>>>(x, xT);
  k_u<<<1536, 256, 0, stream>>>(xT, Wpro, bpro, u);

  dim3 gRow(1, 3072);      // M=NBT, Nc=64  (BN=64)
  dim3 gRow2(1, 3072);     // M=NBT, Nc=128 (BN=128)
  dim3 gGraph(192, 8);     // M=1024, Nc=24576 (BN=128)
  dim3 gL2(8, 8);          // M=1024, Nc=1024  (BN=128)

  // ---- 1x1 convs from x ----
  gemm_bf16_wmma<0, EPI_BIAS_DELAY_BF16, 64><<<gRow, 256, 0, stream>>>(
      xT, wdeT, b_de, delay, nullptr, Db, (int)NBT, 32, 64, 32, 64, 64);
  gemm_bf16_wmma<0, EPI_BIAS_F32, 64><<<gRow, 256, 0, stream>>>(
      xT, w1T, b_1, nullptr, x1f, nullptr, (int)NBT, 32, 64, 32, 64, 64);
  gemm_bf16_wmma<0, EPI_BIAS_BF16, 64><<<gRow, 256, 0, stream>>>(
      xT, wqtT, bq_t, nullptr, nullptr, Qb, (int)NBT, 32, 64, 32, 64, 64);
  gemm_bf16_wmma<0, EPI_BIAS_BF16, 64><<<gRow, 256, 0, stream>>>(
      Db, wktT, bk_t, nullptr, nullptr, Kb, (int)NBT, 64, 64, 64, 64, 64);
  gemm_bf16_wmma<0, EPI_BIAS_BF16, 64><<<gRow, 256, 0, stream>>>(
      Db, wvtT, bv_t, nullptr, nullptr, Vb, (int)NBT, 64, 64, 64, 64, 64);

  // ---- h = u*G + b ; S = (I+adj) @ h ----
  k_h<<<98304, 256, 0, stream>>>(u, G, bmii, hb);
  gemm_bf16_wmma<0, EPI_NONE_BF16, 128><<<gGraph, 256, 0, stream>>>(
      adjI, hb, nullptr, nullptr, nullptr, Sb, 1024, 1024, 24576, 1024, 24576, 24576);

  // ---- attention 1 -> padded XD ----
  (void)hipMemsetAsync(XDp, 0, (size_t)NN * BB * TP * 64 * sizeof(bf16), stream);
  attn_kernel<1, 1><<<2048, 256, 0, stream>>>(Qb, Kb, Vb, x1f, XDp);

  // ---- conv_t as K=192 GEMM over padded windows ----
  gemm_bf16_wmma<1, EPI_BIAS_BF16, 64><<<gRow, 256, 0, stream>>>(
      XDp, wct, b_c1, nullptr, nullptr, x1b, (int)NBT, 192, 64, 0, 64, 64);

  // ---- attention 2 (Q from x1, K/V from S) ----
  gemm_bf16_wmma<0, EPI_BIAS_BF16, 64><<<gRow, 256, 0, stream>>>(
      x1b, wqsT, bq_s, nullptr, nullptr, Qb, (int)NBT, 64, 64, 64, 64, 64);
  gemm_bf16_wmma<0, EPI_BIAS_BF16, 64><<<gRow, 256, 0, stream>>>(
      Sb, wksT, bk_s, nullptr, nullptr, Kb, (int)NBT, 64, 64, 64, 64, 64);
  gemm_bf16_wmma<0, EPI_BIAS_BF16, 64><<<gRow, 256, 0, stream>>>(
      Sb, wvsT, bv_s, nullptr, nullptr, Vb, (int)NBT, 64, 64, 64, 64, 64);
  attn_kernel<0, 0><<<2048, 256, 0, stream>>>(Qb, Kb, Vb, nullptr, XSb);

  // ---- Chebyshev: L2 = 2 sup^2 - I ; Z1 = sup@XS ; Z2 = L2@XS ----
  gemm_bf16_wmma<0, EPI_L2, 128><<<gL2, 256, 0, stream>>>(
      supb, supb, nullptr, nullptr, nullptr, L2b, 1024, 1024, 1024, 1024, 1024, 1024);
  gemm_bf16_wmma<0, EPI_NONE_BF16, 128><<<gGraph, 256, 0, stream>>>(
      supb, XSb, nullptr, nullptr, nullptr, Z1b, 1024, 1024, 24576, 1024, 24576, 24576);
  gemm_bf16_wmma<0, EPI_NONE_BF16, 128><<<gGraph, 256, 0, stream>>>(
      L2b, XSb, nullptr, nullptr, nullptr, Z2b, 1024, 1024, 24576, 1024, 24576, 24576);

  // ---- x2 = XS@Wg0 + Z1@Wg1 + Z2@Wg2 + b_g (f32 accumulate) ----
  gemm_bf16_wmma<0, EPI_BIAS_F32, 128><<<gRow2, 256, 0, stream>>>(
      XSb, wg, b_g, nullptr, x2f, nullptr, (int)NBT, 64, 128, 64, 128, 128);
  gemm_bf16_wmma<0, EPI_ADD_F32, 128><<<gRow2, 256, 0, stream>>>(
      Z1b, wg + 8192, nullptr, nullptr, x2f, nullptr, (int)NBT, 64, 128, 64, 128, 128);
  gemm_bf16_wmma<0, EPI_ADD_F32, 128><<<gRow2, 256, 0, stream>>>(
      Z2b, wg + 16384, nullptr, nullptr, x2f, nullptr, (int)NBT, 64, 128, 64, 128, 128);

  // ---- final gate + transpose to (B, C_OUT, N, T) ----
  k_final<<<98304, 256, 0, stream>>>(x2f, x1f, out);
}
